// DDDhead_13288628814179
// MI455X (gfx1250) — compile-verified, matrix operand order checked by reading, not run on hardware
//
#include <hip/hip_runtime.h>
#include <hip/hip_bf16.h>
#include <math.h>

// ---------------------------------------------------------------------------
// Problem constants
// ---------------------------------------------------------------------------
#define BATCH   4
#define CIN     64
#define HH      96
#define WW      320
#define HWSZ    (HH * WW)          // 30720
#define HC      256
#define NC      3
#define TOPK    100
#define KTOT    (CIN * 9)          // 576
#define KSTEPS  (KTOT / 32)        // 18

// d_out float offsets (tuple: heat, dep, dim, rot, scores, up_pois, reg3d)
#define OFF_HEAT   0ull
#define OFF_DEP    368640ull
#define OFF_DIM    31825920ull
#define OFF_ROT    63283200ull
#define OFF_SCORES 94740480ull
#define OFF_POIS   94740880ull
#define OFF_REG3D  94894480ull

// workspace byte offsets
#define WS_BPACK   0ull            // 4 heads * 147456 ushort  (bf16 packed weights)
#define WS_SS      1179648ull      // 4 heads * 512 float (scale[256], shift[256])
#define WS_NMS     1187840ull      // 368640 float
#define WS_S1S     2662400ull      // 12 * 100 float
#define WS_S1I     2667200ull      // 12 * 100 int
#define WS_S2I     2672000ull      // 4 * 100 int

#define BPACK_PER_HEAD 147456      // 16 ntiles * 18 ksteps * 32 lanes * 16 bf16

typedef __attribute__((ext_vector_type(16))) __bf16 v16bf;
typedef __attribute__((ext_vector_type(8)))  float  v8f;

union ABfrag {
    v16bf v;
    uint4 q[2];
    unsigned short u[16];
};

__device__ __forceinline__ unsigned short f2bf(float f) {
    unsigned int u = __float_as_uint(f);
    u += 0x7FFFu + ((u >> 16) & 1u);           // round-to-nearest-even
    return (unsigned short)(u >> 16);
}

// ---------------------------------------------------------------------------
// Prep: fold bias+BN into scale/shift, repack conv weights into per-lane
// bf16 WMMA B-fragments. K ordering: k = (ky*3+kx)*64 + c.
// Per ISA 16-bit fragment layout, lane's 16 values are two 8-elem K-chunks:
//   k = ks*32 + (j&7) + ((j>>3)<<4) + ((lane>=16)? 8 : 0)
// ---------------------------------------------------------------------------
__global__ __launch_bounds__(256) void pack_kernel(
    const float* __restrict__ w1, const float* __restrict__ b1,
    const float* __restrict__ g,  const float* __restrict__ be,
    const float* __restrict__ mm, const float* __restrict__ vv,
    unsigned short* __restrict__ bpk, float* __restrict__ ss)
{
    int p = blockIdx.x * 256 + threadIdx.x;
    if (p < HC) {
        float inv = g[p] * rsqrtf(vv[p] + 1e-5f);
        ss[p]       = inv;                              // scale
        ss[HC + p]  = (b1[p] - mm[p]) * inv + be[p];    // shift
    }
    if (p >= BPACK_PER_HEAD) return;
    int nt   = p / 9216;          // N tile (16 channels)
    int rem  = p % 9216;
    int ks   = rem / 512;         // K step
    int lr   = rem % 512;
    int lane = lr >> 4;
    int j    = lr & 15;
    int k = ks * 32 + (j & 7) + ((j >> 3) << 4) + ((lane >> 4) << 3);
    int n = nt * 16 + (lane & 15);
    int tap = k >> 6, c = k & 63;
    int ky = tap / 3, kx = tap % 3;
    float val = w1[((n * CIN + c) * 3 + ky) * 3 + kx];  // OIHW
    bpk[p] = f2bf(val);
}

// ---------------------------------------------------------------------------
// Main fused conv3x3 + BN + ReLU (+ hm 1x1 + sigmoid for head 0).
// 512 threads = 16 waves. Block covers 64 consecutive pixels x 256 channels.
// Wave nt owns channels [nt*16, nt*16+16); holds 4 M-tile accumulators.
// ---------------------------------------------------------------------------
__global__ __launch_bounds__(512) void conv_head_kernel(
    const float* __restrict__ up4,
    const unsigned short* __restrict__ Bpack,
    const float* __restrict__ SS,
    const float* __restrict__ hm_w2, const float* __restrict__ hm_b2,
    float* __restrict__ out)
{
    __shared__ __align__(16) unsigned short lds_a[64 * KTOT];   // 73728 B
    __shared__ float hmbuf[64 * HC];                            // 65536 B

    const int head = blockIdx.y;            // 0=hm 1=dep 2=dim 3=rot
    const int m0   = blockIdx.x * 64;
    const int w0   = m0 % WW;
    const int h    = (m0 / WW) % HH;
    const int b    = m0 / (WW * HH);
    const int tid  = threadIdx.x;

    // ---- stage im2col A-tile (64 pixels x 576 K) into LDS as bf16 ----
    const float* inb = up4 + (size_t)b * CIN * HWSZ;
    for (int idx = tid; idx < 64 * KTOT; idx += 512) {
        int m = idx / KTOT, k = idx % KTOT;
        int tap = k >> 6, c = k & 63;
        int ky = tap / 3, kx = tap % 3;
        int hh = h + ky - 1, ww = w0 + m + kx - 1;
        float v = 0.0f;
        if (hh >= 0 && hh < HH && ww >= 0 && ww < WW)
            v = inb[(c * HH + hh) * WW + ww];
        lds_a[idx] = f2bf(v);
    }
    __syncthreads();

    const int lane   = tid & 31;
    const int nt     = tid >> 5;            // wave id == N tile
    const int lhalf  = lane >> 4;           // 0/1 lane half

    v8f acc[4] = {};
    const unsigned short* bp =
        Bpack + (size_t)head * BPACK_PER_HEAD + ((size_t)(nt * KSTEPS) * 32 + lane) * 16;

    for (int ks = 0; ks < KSTEPS; ++ks) {
        ABfrag bf;
        bf.q[0] = *(const uint4*)(bp);
        bf.q[1] = *(const uint4*)(bp + 8);
        bp += 512;                          // next K-step (32 lanes * 16 bf16)
        #pragma unroll
        for (int mtl = 0; mtl < 4; ++mtl) {
            int m = mtl * 16 + (lane & 15);
            const unsigned short* ap = lds_a + m * KTOT + ks * 32 + 8 * lhalf;
            ABfrag af;
            af.q[0] = *(const uint4*)(ap);
            af.q[1] = *(const uint4*)(ap + 16);
            acc[mtl] = __builtin_amdgcn_wmma_f32_16x16x32_bf16(
                false, af.v, false, bf.v, (short)0, acc[mtl], false, false);
        }
    }

    const int ch = nt * 16 + (lane & 15);
    const float scale = SS[head * 512 + ch];
    const float shift = SS[head * 512 + HC + ch];

    if (head == 0) {
        // hm head: keep 256-ch activations in LDS, fuse 1x1 conv + sigmoid
        #pragma unroll
        for (int mtl = 0; mtl < 4; ++mtl)
            #pragma unroll
            for (int i = 0; i < 8; ++i) {
                float r = fmaxf(acc[mtl][i] * scale + shift, 0.0f);
                hmbuf[(mtl * 16 + 8 * lhalf + i) * HC + ch] = r;
            }
        __syncthreads();
        if (tid < 64 * NC) {
            int m = tid & 63, o = tid >> 6;
            float a = hm_b2[o];
            const float* wrow = hm_w2 + o * HC;
            const float* hrow = hmbuf + m * HC;
            #pragma unroll 8
            for (int c = 0; c < HC; ++c) a += wrow[c] * hrow[c];
            float s = 1.0f / (1.0f + __expf(-a));
            out[OFF_HEAT + (((size_t)b * NC + o) * HH + h) * WW + w0 + m] = s;
        }
    } else {
        const size_t ob = (head == 1) ? OFF_DEP : (head == 2) ? OFF_DIM : OFF_ROT;
        #pragma unroll
        for (int mtl = 0; mtl < 4; ++mtl) {
            float4 r0, r1;
            r0.x = fmaxf(acc[mtl][0] * scale + shift, 0.0f);
            r0.y = fmaxf(acc[mtl][1] * scale + shift, 0.0f);
            r0.z = fmaxf(acc[mtl][2] * scale + shift, 0.0f);
            r0.w = fmaxf(acc[mtl][3] * scale + shift, 0.0f);
            r1.x = fmaxf(acc[mtl][4] * scale + shift, 0.0f);
            r1.y = fmaxf(acc[mtl][5] * scale + shift, 0.0f);
            r1.z = fmaxf(acc[mtl][6] * scale + shift, 0.0f);
            r1.w = fmaxf(acc[mtl][7] * scale + shift, 0.0f);
            float* po = out + ob + (((size_t)b * HC + ch) * HH + h) * WW
                        + w0 + mtl * 16 + 8 * lhalf;
            *(float4*)(po)     = r0;
            *(float4*)(po + 4) = r1;
        }
    }
}

// ---------------------------------------------------------------------------
// 3x3 stride-1 max-pool NMS: keep only local maxima
// ---------------------------------------------------------------------------
__global__ __launch_bounds__(256) void nms_kernel(
    const float* __restrict__ heat, float* __restrict__ nmsd)
{
    int idx = blockIdx.x * 256 + threadIdx.x;
    if (idx >= BATCH * NC * HWSZ) return;
    int w = idx % WW, h = (idx / WW) % HH;
    int plane = idx / HWSZ;
    const float* hp = heat + (size_t)plane * HWSZ;
    float v = hp[h * WW + w];
    float mx = -INFINITY;
    #pragma unroll
    for (int dh = -1; dh <= 1; ++dh) {
        int hh = h + dh;
        if (hh < 0 || hh >= HH) continue;
        #pragma unroll
        for (int dw = -1; dw <= 1; ++dw) {
            int ww2 = w + dw;
            if (ww2 < 0 || ww2 >= WW) continue;
            mx = fmaxf(mx, hp[hh * WW + ww2]);
        }
    }
    nmsd[idx] = (v == mx) ? v : 0.0f;
}

// ---------------------------------------------------------------------------
// Top-100 per (b, class): whole 30720-value plane in LDS, iterative argmax
// ---------------------------------------------------------------------------
__global__ __launch_bounds__(256) void topk1_kernel(
    const float* __restrict__ nmsd, float* __restrict__ s1s, int* __restrict__ s1i)
{
    __shared__ float vals[HWSZ];     // 122880 B
    __shared__ float rv[256];
    __shared__ int   ri[256];
    const int bc = blockIdx.x, tid = threadIdx.x;
    const float* src = nmsd + (size_t)bc * HWSZ;
    for (int i = tid; i < HWSZ; i += 256) vals[i] = src[i];
    __syncthreads();
    for (int it = 0; it < TOPK; ++it) {
        float best = -INFINITY; int bi = 0;
        for (int i = tid; i < HWSZ; i += 256) {
            float v = vals[i];
            if (v > best) { best = v; bi = i; }
        }
        rv[tid] = best; ri[tid] = bi;
        __syncthreads();
        for (int s = 128; s > 0; s >>= 1) {
            if (tid < s) {
                if (rv[tid + s] > rv[tid] ||
                    (rv[tid + s] == rv[tid] && ri[tid + s] < ri[tid])) {
                    rv[tid] = rv[tid + s]; ri[tid] = ri[tid + s];
                }
            }
            __syncthreads();
        }
        if (tid == 0) {
            s1s[bc * TOPK + it] = rv[0];
            s1i[bc * TOPK + it] = ri[0];
            vals[ri[0]] = -INFINITY;
        }
        __syncthreads();
    }
}

// ---------------------------------------------------------------------------
// Top-100 of 300 per batch -> final scores + flat indices
// ---------------------------------------------------------------------------
__global__ __launch_bounds__(256) void topk2_kernel(
    const float* __restrict__ s1s, const int* __restrict__ s1i,
    float* __restrict__ scores_out, int* __restrict__ inds_out)
{
    __shared__ float vals[NC * TOPK];
    __shared__ float rv[256];
    __shared__ int   ri[256];
    const int b = blockIdx.x, tid = threadIdx.x;
    for (int i = tid; i < NC * TOPK; i += 256) vals[i] = s1s[b * NC * TOPK + i];
    __syncthreads();
    for (int it = 0; it < TOPK; ++it) {
        float best = -INFINITY; int bi = 0;
        for (int i = tid; i < NC * TOPK; i += 256) {
            float v = vals[i];
            if (v > best) { best = v; bi = i; }
        }
        rv[tid] = best; ri[tid] = bi;
        __syncthreads();
        for (int s = 128; s > 0; s >>= 1) {
            if (tid < s) {
                if (rv[tid + s] > rv[tid] ||
                    (rv[tid + s] == rv[tid] && ri[tid + s] < ri[tid])) {
                    rv[tid] = rv[tid + s]; ri[tid] = ri[tid + s];
                }
            }
            __syncthreads();
        }
        if (tid == 0) {
            int j = ri[0];
            scores_out[b * TOPK + it] = rv[0];
            inds_out[b * TOPK + it]   = s1i[b * NC * TOPK + j];
            vals[j] = -INFINITY;
        }
        __syncthreads();
    }
}

// ---------------------------------------------------------------------------
// Multi-scale feature gather -> up_pois [B, K, 128+256]
// ---------------------------------------------------------------------------
__global__ __launch_bounds__(256) void gather_kernel(
    const float* __restrict__ up8, const float* __restrict__ up16,
    const int* __restrict__ inds, float* __restrict__ out)
{
    int idx = blockIdx.x * 256 + threadIdx.x;
    if (idx >= BATCH * TOPK * 384) return;
    int c = idx % 384;
    int k = (idx / 384) % TOPK;
    int b = idx / (384 * TOPK);
    int ind = inds[b * TOPK + k];
    ind = ind < 0 ? 0 : (ind > HWSZ - 1 ? HWSZ - 1 : ind);
    float v;
    if (c < 128) v = up8[((size_t)b * 128 + c) * HWSZ + (ind >> 1)];
    else         v = up16[((size_t)b * 256 + (c - 128)) * HWSZ + (ind >> 2)];
    out[idx] = v;
}

// ---------------------------------------------------------------------------
// reg3d on the all-ones tensor: value = rowsum(w[o]) + b[o], constant per ch
// ---------------------------------------------------------------------------
__global__ __launch_bounds__(256) void reg3d_kernel(
    const float* __restrict__ w, const float* __restrict__ bias,
    float* __restrict__ out)
{
    __shared__ float s[8];
    if (threadIdx.x < 8) {
        float a = bias[threadIdx.x];
        for (int c = 0; c < 64; ++c) a += w[threadIdx.x * 64 + c];
        s[threadIdx.x] = a;
    }
    __syncthreads();
    int idx = blockIdx.x * 256 + threadIdx.x;
    if (idx >= 8 * 8 * 64 * 320) return;
    int o = (idx / (64 * 320)) & 7;
    out[idx] = s[o];
}

// ---------------------------------------------------------------------------
extern "C" void kernel_launch(void* const* d_in, const int* in_sizes, int n_in,
                              void* d_out, int out_size, void* d_ws, size_t ws_size,
                              hipStream_t stream)
{
    (void)in_sizes; (void)n_in; (void)out_size; (void)ws_size;
    const float* up16  = (const float*)d_in[0];
    const float* up8   = (const float*)d_in[1];
    const float* up4   = (const float*)d_in[2];
    const float* hm_w2 = (const float*)d_in[27];
    const float* hm_b2 = (const float*)d_in[28];
    const float* r3w   = (const float*)d_in[29];
    const float* r3b   = (const float*)d_in[30];
    float* out = (float*)d_out;
    char*  ws  = (char*)d_ws;

    unsigned short* Bpack = (unsigned short*)(ws + WS_BPACK);
    float* SS   = (float*)(ws + WS_SS);
    float* NMSB = (float*)(ws + WS_NMS);
    float* S1S  = (float*)(ws + WS_S1S);
    int*   S1I  = (int*)(ws + WS_S1I);
    int*   S2I  = (int*)(ws + WS_S2I);

    // heads in d_in order: hm(3..8) dep(9..14) dim(15..20) rot(21..26)
    for (int hd = 0; hd < 4; ++hd) {
        int base = 3 + hd * 6;
        pack_kernel<<<BPACK_PER_HEAD / 256, 256, 0, stream>>>(
            (const float*)d_in[base],     (const float*)d_in[base + 1],
            (const float*)d_in[base + 2], (const float*)d_in[base + 3],
            (const float*)d_in[base + 4], (const float*)d_in[base + 5],
            Bpack + (size_t)hd * BPACK_PER_HEAD, SS + hd * 512);
    }

    conv_head_kernel<<<dim3(BATCH * HH * WW / 64, 4, 1), 512, 0, stream>>>(
        up4, Bpack, SS, hm_w2, hm_b2, out);

    nms_kernel<<<(BATCH * NC * HWSZ + 255) / 256, 256, 0, stream>>>(out, NMSB);
    topk1_kernel<<<BATCH * NC, 256, 0, stream>>>(NMSB, S1S, S1I);
    topk2_kernel<<<BATCH, 256, 0, stream>>>(S1S, S1I, out + OFF_SCORES, S2I);
    gather_kernel<<<(BATCH * TOPK * 384 + 255) / 256, 256, 0, stream>>>(
        up8, up16, S2I, out + OFF_POIS);
    reg3d_kernel<<<(8 * 8 * 64 * 320 + 255) / 256, 256, 0, stream>>>(
        r3w, r3b, out + OFF_REG3D);
}
